// EmbeddingLayer_30837865185447
// MI455X (gfx1250) — matrix operand order repaired and checked by reading, not compiled, not run on hardware
//
#include <hip/hip_runtime.h>
#include <hip/hip_bf16.h>

// ---------------- problem constants ----------------
#define BATCH   2048
#define DDIM    150
#define DKDIM   64
#define EXPERT  1774
#define OH_OFF  ((size_t)BATCH * EXPERT)             // 3,633,152
#define FM_OFF  (OH_OFF + (size_t)BATCH * 3)         // 3,639,296

// workspace layout (bytes)
#define WS_X256   ((size_t)0)                                  // 2048*300*4
#define WS_Y256   (WS_X256 + (size_t)BATCH*300*4)              // 2048*320*4
#define WS_X128   (WS_Y256 + (size_t)BATCH*320*4)              // 2048*128*4
#define WS_Y128   (WS_X128 + (size_t)BATCH*128*4)              // 2048*128*4
#define WS_WP256  (WS_Y128 + (size_t)BATCH*128*4)              // 48000*256*2
#define WS_WP128  (WS_WP256 + (size_t)48000*256*2)             // 8192*128*2

typedef __attribute__((ext_vector_type(16))) __bf16        v16bf;
typedef __attribute__((ext_vector_type(8)))  float         v8f;
typedef __attribute__((ext_vector_type(8)))  unsigned int  v8u;

__device__ __forceinline__ unsigned int bf16rne(float f) {
    unsigned int u = __builtin_bit_cast(unsigned int, f);
    return (u + 0x7fffu + ((u >> 16) & 1u)) >> 16;          // round-to-nearest-even
}
// pack two floats to packed-bf16 by truncation: one v_perm_b32
__device__ __forceinline__ unsigned int pack2bf_trunc(float lo, float hi) {
    return __builtin_amdgcn_perm(__builtin_bit_cast(unsigned int, hi),
                                 __builtin_bit_cast(unsigned int, lo),
                                 0x07060302u);
}

// ---------------------------------------------------------------------------
// Kernel 1: gathers + pass-through outputs + staging of GEMM operands
// ---------------------------------------------------------------------------
__global__ __launch_bounds__(256) void gather_kernel(
    const int* __restrict__ inputs, const int* __restrict__ shifts,
    const float* __restrict__ Wuf,  const float* __restrict__ Wfu,
    const float* __restrict__ Wua,  const float* __restrict__ Wau,
    const float* __restrict__ Wfe,
    const float* __restrict__ Wut,  const float* __restrict__ Wuk1,
    const float* __restrict__ Wuk2, const float* __restrict__ Wft,
    const float* __restrict__ Wfk1, const float* __restrict__ Wfk2,
    const float* __restrict__ Wufd, const float* __restrict__ Wfud,
    const float* __restrict__ Wuad, const float* __restrict__ Waud,
    float* __restrict__ out,
    float* __restrict__ X256, float* __restrict__ Y256,
    float* __restrict__ X128, float* __restrict__ Y128)
{
    const int b = blockIdx.x;
    const int t = threadIdx.x;
    const int u = inputs[b * 3 + 0];
    const int f = inputs[b * 3 + 1];
    const int a = inputs[b * 3 + 2];

    float* ex = out + (size_t)b * EXPERT;
    float* fm = out + FM_OFF;

    if (t < 3) out[OH_OFF + (size_t)b * 3 + t] = (float)(inputs[b * 3 + t] + shifts[t]);

    if (t < DDIM) {
        float vuf = Wuf[(size_t)u * DDIM + t];
        ex[t] = vuf;            fm[(size_t)0 * BATCH * DDIM + (size_t)b * DDIM + t] = vuf;
        X256[(size_t)b * 300 + t] = vuf;
        float vua = Wua[(size_t)u * DDIM + t];
        ex[150 + t] = vua;      fm[(size_t)1 * BATCH * DDIM + (size_t)b * DDIM + t] = vua;
        X256[(size_t)b * 300 + 150 + t] = vua;
        float vfu = Wfu[(size_t)f * DDIM + t];
        ex[620 + t] = vfu;      fm[(size_t)2 * BATCH * DDIM + (size_t)b * DDIM + t] = vfu;
        Y256[(size_t)b * 320 + t] = vfu;
        float vau = Wau[(size_t)a * DDIM + t];
        ex[1176 + t] = vau;     fm[(size_t)3 * BATCH * DDIM + (size_t)b * DDIM + t] = vau;
        Y256[(size_t)b * 320 + 160 + t] = vau;
        ex[898 + t] = Wfe[(size_t)f * DDIM + t];
    }
    if (t >= 150 && t < 160) {                       // zero the K padding
        Y256[(size_t)b * 320 + t] = 0.f;
        Y256[(size_t)b * 320 + 160 + t] = 0.f;
    }
    if (t < DKDIM) {
        ex[300 + t]  = Wut[(size_t)u * DKDIM + t];
        ex[364 + t]  = Wuk1[(size_t)u * DKDIM + t];
        ex[428 + t]  = Wuk2[(size_t)u * DKDIM + t];
        float v1 = Wufd[(size_t)u * DKDIM + t];
        ex[492 + t]  = v1;  X128[(size_t)b * 128 + t] = v1;
        float v2 = Wuad[(size_t)u * DKDIM + t];
        ex[556 + t]  = v2;  X128[(size_t)b * 128 + 64 + t] = v2;
        ex[770 + t]  = Wft[(size_t)f * DKDIM + t];
        ex[834 + t]  = Wfk1[(size_t)f * DKDIM + t];
        ex[1048 + t] = Wfk2[(size_t)f * DKDIM + t];
        float v3 = Wfud[(size_t)f * DKDIM + t];
        ex[1112 + t] = v3;  Y128[(size_t)b * 128 + t] = v3;
        float v4 = Waud[(size_t)a * DKDIM + t];
        ex[1326 + t] = v4;  Y128[(size_t)b * 128 + 64 + t] = v4;
    }
}

// ---------------------------------------------------------------------------
// Kernel 2: pack W (f32, K-major [i*JROWS+jp][n]) -> bf16 B-fragment tiling.
//   dst[( (cc*2 + h)*N + n )*16 + idx],  idx = K - 16*h within chunk cc.
// K' = i*JP + jp, jp in [0,JP) zero-padded for jp >= JROWS.  (RNE rounding.)
// ---------------------------------------------------------------------------
template<int JROWS, int JP, int N>
__global__ __launch_bounds__(256) void pack_w(const float* __restrict__ W,
                                              unsigned short* __restrict__ Wp,
                                              int Kp)
{
    size_t tid = (size_t)blockIdx.x * 256 + threadIdx.x;
    int n = (int)(tid % N);
    int k = (int)(tid / N);
    if (k >= Kp) return;
    int cc  = k >> 5;
    int pos = k & 31;
    int hh  = pos >> 4;
    int idx = pos & 15;
    int i  = k / JP;
    int jp = k % JP;
    float v = (jp < JROWS) ? W[(size_t)(i * JROWS + jp) * N + n] : 0.f;
    Wp[((size_t)(cc * 2 + hh) * N + n) * 16 + idx] = (unsigned short)bf16rne(v);
}

// ---------------------------------------------------------------------------
// Kernel 3 helpers: double-buffered chunk pipeline.
// ---------------------------------------------------------------------------
__device__ __forceinline__ void load_btile(const uint4* __restrict__ p, v8u (&dst)[4]) {
    #pragma unroll
    for (int nt = 0; nt < 4; ++nt) {
        uint4 b0 = p[32 * nt];
        uint4 b1 = p[32 * nt + 1];
        dst[nt][0] = b0.x; dst[nt][1] = b0.y; dst[nt][2] = b0.z; dst[nt][3] = b0.w;
        dst[nt][4] = b1.x; dst[nt][5] = b1.y; dst[nt][6] = b1.z; dst[nt][7] = b1.w;
    }
}

// Process all NC chunks of one i.  Buffer parity is compile-time via PBASE:
// chunk (i, c) lives in buf[(c + PBASE) & 1]; while its 4 WMMAs run, the next
// chunk's 8 b128 loads are issued into the other buffer.
template<int NC, int N, int PBASE>
__device__ __forceinline__ void chunk_group(
    const uint4* __restrict__ wp4, unsigned qbase, int ccbase, bool load_next_i,
    float x, const float2 (&yf)[NC][8], v8u (&buf)[2][4], v8f (&acc)[4])
{
    #pragma unroll
    for (int c = 0; c < NC; ++c) {
        const int cur = (c + PBASE) & 1;
        const int nxt = cur ^ 1;
        // issue loads for the next chunk (contiguous chunk ids across i)
        if ((c + 1 < NC) || load_next_i) {
            const uint4* p = wp4 + ((size_t)4 * N * (ccbase + c + 1) + qbase);
            load_btile(p, buf[nxt]);
        }
        // A fragment: x * y, truncate-pack to bf16 (pk_mul + perm per dword)
        v8u au;
        #pragma unroll
        for (int r = 0; r < 8; ++r) {
            float2 y = yf[c][r];
            au[r] = pack2bf_trunc(x * y.x, x * y.y);
        }
        v16bf av = __builtin_bit_cast(v16bf, au);
        #pragma unroll
        for (int nt = 0; nt < 4; ++nt) {
            v16bf bv = __builtin_bit_cast(v16bf, buf[cur][nt]);
            acc[nt] = __builtin_amdgcn_wmma_f32_16x16x32_bf16(
                false, av, false, bv, (short)0, acc[nt], false, false);
        }
    }
}

// ---------------------------------------------------------------------------
// Kernel 3: fused on-the-fly outer-product GEMM + bias + ReLU.
//   out[b, COLOFF + n] = relu( sum_{i,jp} X[b,i]*Ypad[b,jp] * W[i*JP+jp, n] + bias[n] )
// Wave: 16 rows x 64 cols (4 accumulator tiles). Block: 8 waves = 64 x 128.
// y-fragments kept as unpacked float2 in VGPRs, built once per row-tile per
// half; i-loop manually unrolled by 2 so the double-buffer parity stays
// compile-time (PBASE = 0 / NC&1).
// ---------------------------------------------------------------------------
template<int IDIM, int JP, int N, int COLOFF>
__global__ __launch_bounds__(256) void fused_gemm(
    const float* __restrict__ X, const float* __restrict__ Y,
    const unsigned short* __restrict__ Wp, const float* __restrict__ bias,
    float* __restrict__ out)
{
    constexpr int NC = JP / 32;     // K chunks per i
    constexpr int IH = IDIM / 2;    // i per half (even for both GEMMs)
    static_assert((IH & 1) == 0, "IH must be even for 2x i-unroll");
    const int lane   = threadIdx.x & 31;
    const int wave   = threadIdx.x >> 5;
    const int rowgrp = wave >> 1;
    const int colgrp = wave & 1;
    const int row0 = blockIdx.x * 64 + rowgrp * 16;
    const int col0 = blockIdx.y * 128 + colgrp * 64;
    const int m = lane & 15;
    const int h = lane >> 4;
    const int grow = row0 + m;      // A-side source row for this lane

    const uint4* wp4 = (const uint4*)Wp;
    // per-lane B base index (uint4 units); chunk cc adds 4*N*cc, tile nt adds 32*nt
    const unsigned qbase = (unsigned)((h * N + col0 + m) * 2);

    v8f acc[4];
    #pragma unroll
    for (int q = 0; q < 4; ++q) acc[q] = (v8f){0.f,0.f,0.f,0.f,0.f,0.f,0.f,0.f};

    for (int hh = 0; hh < 2; ++hh) {
        // y fragments for this half (unpacked float2), reused for all i
        float2 yf[NC][8];
        const float* yrow = Y + (size_t)grow * (2 * JP) + hh * JP;
        #pragma unroll
        for (int c = 0; c < NC; ++c) {
            #pragma unroll
            for (int r = 0; r < 8; ++r) {
                int jp0 = 32 * c + 16 * h + 2 * r;
                yf[c][r] = *(const float2*)(yrow + jp0);
            }
        }
        const float* xrow  = X + (size_t)grow * IDIM + hh * IH;
        const int    ccb0  = hh * IH * NC;

        // pipeline prologue: first chunk of this half -> buf[0]
        v8u buf[2][4];
        load_btile(wp4 + ((size_t)4 * N * ccb0 + qbase), buf[0]);
        float xcur = xrow[0];

        for (int i2 = 0; i2 < IH / 2; ++i2) {
            const int ii0 = 2 * i2;
            // prefetch two i ahead (clamped to stay in-bounds)
            const int pfii = (ii0 + 2 < IH) ? (ii0 + 2) : 0;
            __builtin_prefetch(
                (const void*)(wp4 + ((size_t)4 * N * (ccb0 + pfii * NC) + qbase)), 0, 1);
            // preload next i's x while this group runs
            float xn0 = xrow[ii0 + 1];
            chunk_group<NC, N, 0>(wp4, qbase, ccb0 + ii0 * NC,
                                  true, xcur, yf, buf, acc);
            const int ii1 = ii0 + 1;
            const bool more = (ii1 + 1 < IH);
            float xn1 = xrow[more ? ii1 + 1 : ii1];
            chunk_group<NC, N, (NC & 1)>(wp4, qbase, ccb0 + ii1 * NC,
                                         more, xn0, yf, buf, acc);
            xcur = xn1;
        }
    }
    // epilogue: bias + relu, scatter into expert_inputs columns
    #pragma unroll
    for (int nt = 0; nt < 4; ++nt) {
        int n = col0 + nt * 16 + m;
        float bsv = bias[n];
        #pragma unroll
        for (int r = 0; r < 8; ++r) {
            int row = row0 + h * 8 + r;      // C/D layout: lanes16-31 hold M = 8+r
            float v = acc[nt][r] + bsv;
            v = v > 0.f ? v : 0.f;
            out[(size_t)row * EXPERT + COLOFF + n] = v;
        }
    }
}

// ---------------------------------------------------------------------------
extern "C" void kernel_launch(void* const* d_in, const int* in_sizes, int n_in,
                              void* d_out, int out_size, void* d_ws, size_t ws_size,
                              hipStream_t stream)
{
    const int*   inputs = (const int*)d_in[0];
    const int*   shifts = (const int*)d_in[1];
    const float* Wuf  = (const float*)d_in[2];
    const float* Wfu  = (const float*)d_in[3];
    const float* Wua  = (const float*)d_in[4];
    const float* Wau  = (const float*)d_in[5];
    const float* Wfe  = (const float*)d_in[6];
    const float* Wut  = (const float*)d_in[7];
    const float* Wuk1 = (const float*)d_in[8];
    const float* Wuk2 = (const float*)d_in[9];
    const float* Wft  = (const float*)d_in[10];
    const float* Wfk1 = (const float*)d_in[11];
    const float* Wfk2 = (const float*)d_in[12];
    const float* Wufd = (const float*)d_in[13];
    const float* Wfud = (const float*)d_in[14];
    const float* Wuad = (const float*)d_in[15];
    const float* Waud = (const float*)d_in[16];
    const float* W256 = (const float*)d_in[17];
    const float* b256 = (const float*)d_in[18];
    const float* W128 = (const float*)d_in[19];
    const float* b128 = (const float*)d_in[20];

    float* out = (float*)d_out;
    char*  ws  = (char*)d_ws;
    float* X256 = (float*)(ws + WS_X256);
    float* Y256 = (float*)(ws + WS_Y256);
    float* X128 = (float*)(ws + WS_X128);
    float* Y128 = (float*)(ws + WS_Y128);
    unsigned short* Wp256 = (unsigned short*)(ws + WS_WP256);
    unsigned short* Wp128 = (unsigned short*)(ws + WS_WP128);

    gather_kernel<<<BATCH, 256, 0, stream>>>(
        inputs, shifts, Wuf, Wfu, Wua, Wau, Wfe, Wut, Wuk1, Wuk2,
        Wft, Wfk1, Wfk2, Wufd, Wfud, Wuad, Waud,
        out, X256, Y256, X128, Y128);

    // pack weights to bf16 fragment tiling (K' = 300*160 = 48000 ; 128*64 = 8192)
    pack_w<150, 160, 256><<<48000, 256, 0, stream>>>(W256, Wp256, 48000);
    pack_w< 64,  64, 128><<< 4096, 256, 0, stream>>>(W128, Wp128, 8192);

    // user_x_user -> expert cols [1390,1646)
    fused_gemm<300, 160, 256, 1390><<<dim3(BATCH / 64, 2), 256, 0, stream>>>(
        X256, Y256, Wp256, b256, out);
    // user_w_user -> expert cols [1646,1774)
    fused_gemm<128,  64, 128, 1646><<<dim3(BATCH / 64, 1), 256, 0, stream>>>(
        X128, Y128, Wp128, b128, out);
}
